// SelfAttention_87960930222705
// MI455X (gfx1250) — matrix operand order repaired and checked by reading, not compiled, not run on hardware
//
#include <hip/hip_runtime.h>
#include <hip/hip_bf16.h>
#include <math.h>

typedef _Float16 half_t;
typedef __attribute__((ext_vector_type(16))) _Float16 v16h;
typedef __attribute__((ext_vector_type(8)))  float    v8f;

#define D_MODEL 1024
#define N_HEADS 16
#define D_K     64
#define SEQ     2048
#define BATCH   4
#define ROWS    (BATCH * SEQ)   // 8192

// ---------------------------------------------------------------------------
// WMMA fragment helpers (wave32, v_wmma_f32_16x16x32_f16)
//
// A fragment (16x32 f16, row-major source, stride lda):
//   lane l: m = l&15, h = l>>4; half j: K = j + (j>=8 ? 8 : 0) + 8*h
// B fragment loaded via B-transpose (row-major [N,K], stride ldb):
//   lane l: n = l&15, h = l>>4; half j: K = k0 + j + (j>=8 ? 8 : 0) + 8*h
// C/D f32: lane l: n = l&15; vgpr r: m = r + 8*(l>>4)
// ---------------------------------------------------------------------------

__device__ __forceinline__ v16h load_A_frag(const half_t* A, int lda, int m0, int k0) {
    int lane = threadIdx.x & 31;
    int m = m0 + (lane & 15);
    int h = lane >> 4;
    const half_t* row = A + (size_t)m * lda + k0 + 8 * h;
    v16h a;
#pragma unroll
    for (int p = 0; p < 8; ++p) {
        int kk = 2 * p + (p >= 4 ? 8 : 0);
        a[2 * p]     = row[kk];
        a[2 * p + 1] = row[kk + 1];
    }
    return a;
}

__device__ __forceinline__ v16h load_Bt_frag(const half_t* Bt, int ldb, int n0, int k0) {
    int lane = threadIdx.x & 31;
    int n = n0 + (lane & 15);
    int h = lane >> 4;
    const half_t* row = Bt + (size_t)n * ldb + k0 + 8 * h;
    v16h b;
#pragma unroll
    for (int j = 0; j < 16; ++j) b[j] = row[j + (j >= 8 ? 8 : 0)];
    return b;
}

__device__ __forceinline__ v8f wmma_f16(v16h a, v16h b, v8f c) {
    return __builtin_amdgcn_wmma_f32_16x16x32_f16(false, a, false, b, (short)0, c,
                                                  false, false);
}

// ---------------------------------------------------------------------------
// K0a: convert + transpose per-head projection weights [H,D,DK]f32 -> [H,DK,D]f16
// ---------------------------------------------------------------------------
__global__ void convW_kernel(const float* __restrict__ W, half_t* __restrict__ Wt) {
    int idx = blockIdx.x * 256 + threadIdx.x;      // over H*D*DK = 1,048,576
    int h   = idx >> 16;                           // / (D_MODEL*D_K)
    int rem = idx & 65535;
    int d   = rem >> 6;
    int k   = rem & 63;
    Wt[((size_t)h * D_K + k) * D_MODEL + d] = (half_t)W[idx];
}

// K0b: plain f32 -> f16 convert (Wo stays [dout,din] = transposed for X*Wo^T)
__global__ void convWo_kernel(const float* __restrict__ W, half_t* __restrict__ Wt) {
    int idx = blockIdx.x * 256 + threadIdx.x;
    Wt[idx] = (half_t)W[idx];
}

// ---------------------------------------------------------------------------
// K1: LayerNorm (no affine) -> f16. One 256-thread block per row.
// ---------------------------------------------------------------------------
__global__ void ln_kernel(const float* __restrict__ x, half_t* __restrict__ xn) {
    int row = blockIdx.x;
    const float* xr = x + (size_t)row * D_MODEL;
    __shared__ float red[256];

    float s = 0.0f;
    for (int i = threadIdx.x; i < D_MODEL; i += 256) s += xr[i];
    red[threadIdx.x] = s;
    __syncthreads();
    for (int off = 128; off > 0; off >>= 1) {
        if (threadIdx.x < off) red[threadIdx.x] += red[threadIdx.x + off];
        __syncthreads();
    }
    float mu = red[0] * (1.0f / D_MODEL);
    __syncthreads();

    float v = 0.0f;
    for (int i = threadIdx.x; i < D_MODEL; i += 256) {
        float d = xr[i] - mu;
        v += d * d;
    }
    red[threadIdx.x] = v;
    __syncthreads();
    for (int off = 128; off > 0; off >>= 1) {
        if (threadIdx.x < off) red[threadIdx.x] += red[threadIdx.x + off];
        __syncthreads();
    }
    float rstd = rsqrtf(red[0] * (1.0f / D_MODEL) + 1e-5f);

    half_t* o = xn + (size_t)row * D_MODEL;
    for (int i = threadIdx.x; i < D_MODEL; i += 256)
        o[i] = (half_t)((xr[i] - mu) * rstd);
}

// ---------------------------------------------------------------------------
// K2: QKV projection. grid = (ROWS/16, H, 3), block = 32 (one wave).
// One wave computes a 16x64 tile (full head width): 1 A-frag -> 4 WMMAs.
// All 4 B-frags hoisted into distinct registers so loads clause together and
// the 4 WMMAs issue back-to-back.
// Q,K -> [B,H,S,64] f16 ; V -> transposed [B,H,64,S] f16
// ---------------------------------------------------------------------------
__global__ void qkv_kernel(const half_t* __restrict__ xn,
                           const half_t* __restrict__ WtQ,
                           const half_t* __restrict__ WtK,
                           const half_t* __restrict__ WtV,
                           const float* __restrict__ bq,
                           const float* __restrict__ bk,
                           const float* __restrict__ bv,
                           half_t* __restrict__ Qb,
                           half_t* __restrict__ Kb,
                           half_t* __restrict__ Vt) {
    int m0   = blockIdx.x * 16;
    int hidx = blockIdx.y;
    int mat  = blockIdx.z;
    const half_t* Wt   = (mat == 0) ? WtQ : (mat == 1) ? WtK : WtV;
    const float*  bias = (mat == 0) ? bq  : (mat == 1) ? bk  : bv;
    const half_t* Wth  = Wt + (size_t)hidx * D_K * D_MODEL;

    v8f acc[4] = {};
#pragma unroll 2
    for (int k0 = 0; k0 < D_MODEL; k0 += 32) {
        v16h a  = load_A_frag(xn, D_MODEL, m0, k0);
        v16h b0 = load_Bt_frag(Wth, D_MODEL, 0,  k0);
        v16h b1 = load_Bt_frag(Wth, D_MODEL, 16, k0);
        v16h b2 = load_Bt_frag(Wth, D_MODEL, 32, k0);
        v16h b3 = load_Bt_frag(Wth, D_MODEL, 48, k0);
        const half_t* pf =
            xn + (size_t)(m0 + (threadIdx.x & 15)) * D_MODEL + k0 + 64;
        __builtin_prefetch(pf, 0, 3);
        acc[0] = wmma_f16(a, b0, acc[0]);
        acc[1] = wmma_f16(a, b1, acc[1]);
        acc[2] = wmma_f16(a, b2, acc[2]);
        acc[3] = wmma_f16(a, b3, acc[3]);
    }

    int lane = threadIdx.x & 31;
    int n  = lane & 15;
    int hh = lane >> 4;

#pragma unroll
    for (int t = 0; t < 4; ++t) {
        int dk = t * 16 + n;
        float bval = bias[hidx * D_K + dk];
        if (mat == 2) {
#pragma unroll
            for (int r = 0; r < 8; ++r) {
                int row = m0 + r + 8 * hh;
                int b_  = row >> 11;      // / SEQ
                int s   = row & (SEQ - 1);
                Vt[(((size_t)b_ * N_HEADS + hidx) * D_K + dk) * SEQ + s] =
                    (half_t)(acc[t][r] + bval);
            }
        } else {
            half_t* Out = (mat == 0) ? Qb : Kb;
#pragma unroll
            for (int r = 0; r < 8; ++r) {
                int row = m0 + r + 8 * hh;
                int b_  = row >> 11;
                int s   = row & (SEQ - 1);
                Out[(((size_t)b_ * N_HEADS + hidx) * SEQ + s) * D_K + dk] =
                    (half_t)(acc[t][r] + bval);
            }
        }
    }
}

// ---------------------------------------------------------------------------
// K3: column-softmax stats. softmax is over the QUERY axis, so per key s we
// need m[s] = max_q score/8, l[s] = sum_q exp(score/8 - m[s]).
// grid = (SEQ/32, H, B), block = 32. One wave owns a 32-wide key strip:
// 4 resident K-fragments reused across all q-tiles; each A-frag feeds 2 WMMAs.
// ---------------------------------------------------------------------------
__global__ void stats_kernel(const half_t* __restrict__ Qb,
                             const half_t* __restrict__ Kb,
                             float* __restrict__ Sm,
                             float* __restrict__ Sl) {
    int st   = blockIdx.x;           // 32-key strip
    int hidx = blockIdx.y;
    int b    = blockIdx.z;
    size_t bh = (size_t)b * N_HEADS + hidx;
    const half_t* Q = Qb + bh * SEQ * D_K;
    const half_t* K = Kb + bh * SEQ * D_K;

    v16h bk00 = load_Bt_frag(K, D_K, st * 32,      0);
    v16h bk01 = load_Bt_frag(K, D_K, st * 32,      32);
    v16h bk10 = load_Bt_frag(K, D_K, st * 32 + 16, 0);
    v16h bk11 = load_Bt_frag(K, D_K, st * 32 + 16, 32);

    const float scale = 0.125f;   // 1/sqrt(64)
    float m_run[2] = {-3.0e38f, -3.0e38f};
    float l_run[2] = {0.0f, 0.0f};

    for (int qt = 0; qt < SEQ / 16; ++qt) {
        v16h a0 = load_A_frag(Q, D_K, qt * 16, 0);
        v16h a1 = load_A_frag(Q, D_K, qt * 16, 32);
#pragma unroll
        for (int u = 0; u < 2; ++u) {
            v8f c = {};
            c = wmma_f16(a0, (u == 0) ? bk00 : bk10, c);
            c = wmma_f16(a1, (u == 0) ? bk01 : bk11, c);

            float tmax = c[0];
#pragma unroll
            for (int r = 1; r < 8; ++r) tmax = fmaxf(tmax, c[r]);
            tmax *= scale;
            float m_new = fmaxf(m_run[u], tmax);
            float ssum = 0.0f;
#pragma unroll
            for (int r = 0; r < 8; ++r) ssum += __expf(c[r] * scale - m_new);
            l_run[u] = l_run[u] * __expf(m_run[u] - m_new) + ssum;
            m_run[u] = m_new;
        }
    }

    int lane = threadIdx.x & 31;
#pragma unroll
    for (int u = 0; u < 2; ++u) {
        // column n is split across lanes l and l+16 (8 rows each) -> merge
        float om = __shfl_xor(m_run[u], 16, 32);
        float ol = __shfl_xor(l_run[u], 16, 32);
        float mf = fmaxf(m_run[u], om);
        float lf = l_run[u] * __expf(m_run[u] - mf) + ol * __expf(om - mf);
        if (lane < 16) {
            int s = st * 32 + u * 16 + lane;
            Sm[bh * SEQ + s] = mf;
            Sl[bh * SEQ + s] = lf;
        }
    }
}

// ---------------------------------------------------------------------------
// K4: second pass: heads[q,v] = sum_s exp(score/8 - m[s])/l[s] * V[s,v].
// grid = (SEQ/32, H, B), block = 32. One wave handles TWO q-tiles so each
// K-fragment and V-fragment feeds two WMMAs. Per 32-key step: 8 score WMMAs,
// LDS transpose of the two 16x32 weight tiles, 8 accumulate WMMAs.
// ---------------------------------------------------------------------------
__global__ void attn_kernel(const half_t* __restrict__ Qb,
                            const half_t* __restrict__ Kb,
                            const half_t* __restrict__ Vt,
                            const float* __restrict__ Sm,
                            const float* __restrict__ Sl,
                            half_t* __restrict__ Hd) {
    int qt   = blockIdx.x;           // 32-query strip (2 tiles)
    int hidx = blockIdx.y;
    int b    = blockIdx.z;
    size_t bh = (size_t)b * N_HEADS + hidx;
    const half_t* Q  = Qb + bh * SEQ * D_K;
    const half_t* K  = Kb + bh * SEQ * D_K;
    const half_t* V  = Vt + bh * D_K * SEQ;
    const float*  sm = Sm + bh * SEQ;
    const float*  sl = Sl + bh * SEQ;

    __shared__ half_t wt[2][16 * 32];

    v16h aq[2][2];
#pragma unroll
    for (int qu = 0; qu < 2; ++qu) {
        aq[qu][0] = load_A_frag(Q, D_K, qt * 32 + qu * 16, 0);
        aq[qu][1] = load_A_frag(Q, D_K, qt * 32 + qu * 16, 32);
    }
    v8f acc[2][4] = {};

    int lane = threadIdx.x & 31;
    int n  = lane & 15;
    int hh = lane >> 4;
    const float scale = 0.125f;

    for (int s0 = 0; s0 < SEQ; s0 += 32) {
#pragma unroll
        for (int u = 0; u < 2; ++u) {
            int sb = s0 + 16 * u;
            v16h bk0 = load_Bt_frag(K, D_K, sb, 0);
            v16h bk1 = load_Bt_frag(K, D_K, sb, 32);
            int s = sb + n;
            float ms  = sm[s];
            float inv = 1.0f / sl[s];
#pragma unroll
            for (int qu = 0; qu < 2; ++qu) {
                v8f c = {};
                c = wmma_f16(aq[qu][0], bk0, c);
                c = wmma_f16(aq[qu][1], bk1, c);
#pragma unroll
                for (int r = 0; r < 8; ++r) {
                    float w = __expf(c[r] * scale - ms) * inv;
                    wt[qu][(r + 8 * hh) * 32 + 16 * u + n] = (half_t)w;
                }
            }
        }
        __syncthreads();

        v16h aw0 = load_A_frag(wt[0], 32, 0, 0);
        v16h aw1 = load_A_frag(wt[1], 32, 0, 0);
        v16h bv0 = load_Bt_frag(V, SEQ, 0,  s0);
        v16h bv1 = load_Bt_frag(V, SEQ, 16, s0);
        v16h bv2 = load_Bt_frag(V, SEQ, 32, s0);
        v16h bv3 = load_Bt_frag(V, SEQ, 48, s0);
        acc[0][0] = wmma_f16(aw0, bv0, acc[0][0]);
        acc[1][0] = wmma_f16(aw1, bv0, acc[1][0]);
        acc[0][1] = wmma_f16(aw0, bv1, acc[0][1]);
        acc[1][1] = wmma_f16(aw1, bv1, acc[1][1]);
        acc[0][2] = wmma_f16(aw0, bv2, acc[0][2]);
        acc[1][2] = wmma_f16(aw1, bv2, acc[1][2]);
        acc[0][3] = wmma_f16(aw0, bv3, acc[0][3]);
        acc[1][3] = wmma_f16(aw1, bv3, acc[1][3]);
        __syncthreads();
    }

    size_t rowbase = (size_t)b * SEQ;
#pragma unroll
    for (int qu = 0; qu < 2; ++qu) {
#pragma unroll
        for (int r = 0; r < 8; ++r) {
            int q = qt * 32 + qu * 16 + r + 8 * hh;
            half_t* o = Hd + (rowbase + q) * D_MODEL + hidx * D_K;
            o[0  + n] = (half_t)acc[qu][0][r];
            o[16 + n] = (half_t)acc[qu][1][r];
            o[32 + n] = (half_t)acc[qu][2][r];
            o[48 + n] = (half_t)acc[qu][3][r];
        }
    }
}

// ---------------------------------------------------------------------------
// K5: output projection + bias + residual (f32 out).
// out[row,n] = sum_k heads[row,k]*Wo[n,k] + bo[n] + x[row,n]
// grid = (ROWS/16, D_MODEL/64), block = 32. 16x64 tile: 1 A-frag -> 4 WMMAs.
// ---------------------------------------------------------------------------
__global__ void outproj_kernel(const half_t* __restrict__ Hd,
                               const half_t* __restrict__ Wo16,
                               const float* __restrict__ bo,
                               const float* __restrict__ x,
                               float* __restrict__ out) {
    int m0 = blockIdx.x * 16;
    int n0 = blockIdx.y * 64;
    v8f acc[4] = {};
#pragma unroll 2
    for (int k0 = 0; k0 < D_MODEL; k0 += 32) {
        v16h a  = load_A_frag(Hd, D_MODEL, m0, k0);
        v16h b0 = load_Bt_frag(Wo16, D_MODEL, n0,      k0);
        v16h b1 = load_Bt_frag(Wo16, D_MODEL, n0 + 16, k0);
        v16h b2 = load_Bt_frag(Wo16, D_MODEL, n0 + 32, k0);
        v16h b3 = load_Bt_frag(Wo16, D_MODEL, n0 + 48, k0);
        const half_t* pf =
            Hd + (size_t)(m0 + (threadIdx.x & 15)) * D_MODEL + k0 + 64;
        __builtin_prefetch(pf, 0, 3);
        acc[0] = wmma_f16(a, b0, acc[0]);
        acc[1] = wmma_f16(a, b1, acc[1]);
        acc[2] = wmma_f16(a, b2, acc[2]);
        acc[3] = wmma_f16(a, b3, acc[3]);
    }
    int lane = threadIdx.x & 31;
    int hh = lane >> 4;
#pragma unroll
    for (int t = 0; t < 4; ++t) {
        int n = n0 + t * 16 + (lane & 15);
        float bb = bo[n];
#pragma unroll
        for (int r = 0; r < 8; ++r) {
            size_t row = (size_t)(m0 + r + 8 * hh);
            out[row * D_MODEL + n] = acc[t][r] + bb + x[row * D_MODEL + n];
        }
    }
}

// ---------------------------------------------------------------------------
extern "C" void kernel_launch(void* const* d_in, const int* in_sizes, int n_in,
                              void* d_out, int out_size, void* d_ws, size_t ws_size,
                              hipStream_t stream) {
    const float* x  = (const float*)d_in[0];
    const float* Wq = (const float*)d_in[1];
    const float* bq = (const float*)d_in[2];
    const float* Wk = (const float*)d_in[3];
    const float* bk = (const float*)d_in[4];
    const float* Wv = (const float*)d_in[5];
    const float* bv = (const float*)d_in[6];
    const float* Wo = (const float*)d_in[7];
    const float* bo = (const float*)d_in[8];
    float* out = (float*)d_out;

    // workspace carve-up (all sizes multiples of 4KB)
    char* p = (char*)d_ws;
    half_t* xn   = (half_t*)p; p += (size_t)ROWS * D_MODEL * 2;              // 16 MB
    half_t* WtQ  = (half_t*)p; p += (size_t)N_HEADS * D_K * D_MODEL * 2;     //  2 MB
    half_t* WtK  = (half_t*)p; p += (size_t)N_HEADS * D_K * D_MODEL * 2;     //  2 MB
    half_t* WtV  = (half_t*)p; p += (size_t)N_HEADS * D_K * D_MODEL * 2;     //  2 MB
    half_t* Wo16 = (half_t*)p; p += (size_t)D_MODEL * D_MODEL * 2;           //  2 MB
    half_t* Qb   = (half_t*)p; p += (size_t)BATCH * N_HEADS * SEQ * D_K * 2; // 16 MB
    half_t* Kb   = (half_t*)p; p += (size_t)BATCH * N_HEADS * SEQ * D_K * 2; // 16 MB
    half_t* Vt   = (half_t*)p; p += (size_t)BATCH * N_HEADS * SEQ * D_K * 2; // 16 MB
    float*  Sm   = (float*)p;  p += (size_t)BATCH * N_HEADS * SEQ * 4;       // 0.5 MB
    float*  Sl   = (float*)p;  p += (size_t)BATCH * N_HEADS * SEQ * 4;       // 0.5 MB
    half_t* Hd   = (half_t*)p; p += (size_t)ROWS * D_MODEL * 2;              // 16 MB

    // weight conversion / transposition
    convW_kernel<<<dim3(4096), dim3(256), 0, stream>>>(Wq, WtQ);
    convW_kernel<<<dim3(4096), dim3(256), 0, stream>>>(Wk, WtK);
    convW_kernel<<<dim3(4096), dim3(256), 0, stream>>>(Wv, WtV);
    convWo_kernel<<<dim3(4096), dim3(256), 0, stream>>>(Wo, Wo16);

    // layernorm
    ln_kernel<<<dim3(ROWS), dim3(256), 0, stream>>>(x, xn);

    // QKV projection
    qkv_kernel<<<dim3(ROWS / 16, N_HEADS, 3), dim3(32), 0, stream>>>(
        xn, WtQ, WtK, WtV, bq, bk, bv, Qb, Kb, Vt);

    // column-softmax stats (softmax over query axis)
    stats_kernel<<<dim3(SEQ / 32, N_HEADS, BATCH), dim3(32), 0, stream>>>(
        Qb, Kb, Sm, Sl);

    // attention second pass -> heads
    attn_kernel<<<dim3(SEQ / 32, N_HEADS, BATCH), dim3(32), 0, stream>>>(
        Qb, Kb, Vt, Sm, Sl, Hd);

    // output projection + residual
    outproj_kernel<<<dim3(ROWS / 16, D_MODEL / 64), dim3(32), 0, stream>>>(
        Hd, Wo16, bo, x, out);
}
